// Encoder_26087631356042
// MI455X (gfx1250) — matrix operand order repaired and checked by reading, not compiled, not run on hardware
//
#include <hip/hip_runtime.h>

#define B_  128
#define T_  2048
#define X_  128
#define H_  256
#define G3  768

#define WIH_HALVES (G3 * X_)   // 98304  f16 elements of swizzled W_ih
#define WHH_HALVES (G3 * H_)   // 196608 f16 elements of swizzled W_hh
#define HP 264                 // padded f16 row stride for LDS h buffer

typedef _Float16 v16h __attribute__((ext_vector_type(16)));
typedef _Float16 v8h  __attribute__((ext_vector_type(8)));
typedef float    v8f  __attribute__((ext_vector_type(8)));
typedef float    v4f  __attribute__((ext_vector_type(4)));

// ---------------------------------------------------------------------------
// Prep: swizzle f32 weights into f16 WMMA B-fragment layout in workspace.
// B-fragment (16x16x32 f16): lane l holds column n = l&15; halves i=0..15
// hold k = (i>=8 ? 16 : 0) + (l>=16 ? 8 : 0) + (i&7) of the 32-wide K chunk. One fragment =
// 32 lanes * 16 halves = 1024 B, stored contiguously (lane-major).
// ---------------------------------------------------------------------------
__global__ void gru_prep(const float* __restrict__ W_ih,
                         const float* __restrict__ W_hh,
                         _Float16* __restrict__ ws16) {
  int idx = blockIdx.x * blockDim.x + threadIdx.x;
  const int total = WIH_HALVES + WHH_HALVES;
  if (idx >= total) return;
  if (idx < WIH_HALVES) {
    int frag   = idx >> 9;          // j*4 + kc   (j = gate-row chunk 0..47)
    int within = idx & 511;
    int lane   = within >> 4;
    int i      = within & 15;
    int n = (frag >> 2) * 16 + (lane & 15);
    int k = (frag & 3) * 32 + ((i >> 3) << 4) + ((lane >> 4) << 3) + (i & 7);
    ws16[idx] = (_Float16)W_ih[n * X_ + k];
  } else {
    int idx2   = idx - WIH_HALVES;
    int frag   = idx2 >> 9;         // j*8 + kc
    int within = idx2 & 511;
    int lane   = within >> 4;
    int i      = within & 15;
    int n = (frag >> 3) * 16 + (lane & 15);
    int k = (frag & 7) * 32 + ((i >> 3) << 4) + ((lane >> 4) << 3) + (i & 7);
    ws16[idx] = (_Float16)W_hh[n * H_ + k];
  }
}

// ---------------------------------------------------------------------------
// Fused GRU scan. Grid = 8 blocks (batch tiles of 16 rows), block = 512
// threads = 16 waves. Wave w owns H columns [16w,16w+16) and its r/z/n gate
// tiles. Hidden state carried in f32 registers; f16 copy in double-buffered
// LDS for WMMA A-fragments. One barrier per time step. Weight B-fragments are
// streamed from the L2-resident swizzled workspace each step; an opaque-asm
// ZERO OFFSET (not an opaque pointer — that would lose the global address
// space and force flat_load) makes the loads loop-variant so LICM cannot
// hoist + spill them, while they remain global_load_b128.
// ---------------------------------------------------------------------------
__global__ __launch_bounds__(512)
void gru_scan(const float* __restrict__ input,
              const float* __restrict__ b_ih,
              const float* __restrict__ b_hh,
              const _Float16* __restrict__ ws16,
              float* __restrict__ out) {
  __shared__ _Float16 hbuf[2][16 * HP];

  const int tid  = threadIdx.x;
  const int w    = tid >> 5;        // wave id 0..15
  const int lane = tid & 31;
  const int lnib = lane & 15;
  const int hi   = lane >> 4;       // lane half
  const int b0   = blockIdx.x * 16; // batch tile base

  // zero the first h buffer (h0 = 0)
  for (int i = tid; i < 16 * HP; i += 512) hbuf[0][i] = (_Float16)0.0f;

  // Per-lane fragment base pointers (lane offset folded in once).
  const _Float16* wihp = ws16 + ((size_t)lane << 4);
  const _Float16* whhp = ws16 + WIH_HALVES + ((size_t)lane << 4);

  // Biases for this lane's output columns
  float bi[3], bh[3];
  #pragma unroll
  for (int g = 0; g < 3; ++g) {
    int col = g * H_ + w * 16 + lnib;
    bi[g] = b_ih[col];
    bh[g] = b_hh[col];
  }

  // A-fragment K offset within a 32-chunk for this lane half
  const int kbA = hi ? 8 : 0;
  // input row for A-fragment row m = lnib
  const float* in_row = input + (size_t)(b0 + lnib) * T_ * X_;

  float hreg[8];
  #pragma unroll
  for (int v = 0; v < 8; ++v) hreg[v] = 0.0f;

  __syncthreads();

  for (int t = 0; t < T_; ++t) {
    const _Float16* rb = hbuf[t & 1];
    _Float16*       wb = hbuf[(t & 1) ^ 1];

    // Opaque zero offset: redefined by volatile asm every iteration, so the
    // weight-fragment loads are loop-variant (no LICM hoist -> no spill), yet
    // the pointer chain stays rooted at the global kernel arg, keeping the
    // accesses as global_load_b128 (DScnt stays reserved for real LDS ops).
    unsigned zoff = 0;
    asm volatile("" : "+v"(zoff));
    const _Float16* wih_t = wihp + zoff;
    const _Float16* whh_t = whhp + zoff;

    v8f acc_r = {}, acc_z = {}, acc_xn = {}, acc_hn = {};

    // ---- recurrent part first: its LDS loads are the only barrier-dependent
    // ones, so issue them ASAP.  hp = h @ W_hh^T (K = 256, 8 chunks of 32) ----
    #pragma unroll
    for (int kc = 0; kc < 8; ++kc) {
      v8h lo  = *(const v8h*)(rb + lnib * HP + kc * 32 + kbA);
      v8h hi8 = *(const v8h*)(rb + lnib * HP + kc * 32 + 16 + kbA);
      v16h a  = __builtin_shufflevector(lo, hi8, 0,1,2,3,4,5,6,7,8,9,10,11,12,13,14,15);
      v16h br = *(const v16h*)(whh_t + ((size_t)((0 * 16 + w) * 8 + kc) << 9));
      v16h bz = *(const v16h*)(whh_t + ((size_t)((1 * 16 + w) * 8 + kc) << 9));
      v16h bn = *(const v16h*)(whh_t + ((size_t)((2 * 16 + w) * 8 + kc) << 9));
      acc_r  = __builtin_amdgcn_wmma_f32_16x16x32_f16(false, a, false, br, (short)0, acc_r,  false, false);
      acc_z  = __builtin_amdgcn_wmma_f32_16x16x32_f16(false, a, false, bz, (short)0, acc_z,  false, false);
      acc_hn = __builtin_amdgcn_wmma_f32_16x16x32_f16(false, a, false, bn, (short)0, acc_hn, false, false);
    }

    // ---- input projection part (independent of h; overlaps with the WMMA
    // chain above) ----
    const float* xin = in_row + (size_t)t * X_;
    #pragma unroll
    for (int kc = 0; kc < 4; ++kc) {
      v4f f0 = *(const v4f*)(xin + kc * 32 + kbA);
      v4f f1 = *(const v4f*)(xin + kc * 32 + kbA + 4);
      v4f f2 = *(const v4f*)(xin + kc * 32 + 16 + kbA);
      v4f f3 = *(const v4f*)(xin + kc * 32 + 16 + kbA + 4);
      v16h a;
      #pragma unroll
      for (int j = 0; j < 4; ++j) {
        a[j]      = (_Float16)f0[j];
        a[4 + j]  = (_Float16)f1[j];
        a[8 + j]  = (_Float16)f2[j];
        a[12 + j] = (_Float16)f3[j];
      }
      v16h bxr = *(const v16h*)(wih_t + ((size_t)((0 * 16 + w) * 4 + kc) << 9));
      v16h bxz = *(const v16h*)(wih_t + ((size_t)((1 * 16 + w) * 4 + kc) << 9));
      v16h bxn = *(const v16h*)(wih_t + ((size_t)((2 * 16 + w) * 4 + kc) << 9));
      acc_r  = __builtin_amdgcn_wmma_f32_16x16x32_f16(false, a, false, bxr, (short)0, acc_r,  false, false);
      acc_z  = __builtin_amdgcn_wmma_f32_16x16x32_f16(false, a, false, bxz, (short)0, acc_z,  false, false);
      acc_xn = __builtin_amdgcn_wmma_f32_16x16x32_f16(false, a, false, bxn, (short)0, acc_xn, false, false);
    }

    // ---- gates + state update (D layout: row m = v + 8*hi, col = 16w+lnib)
    #pragma unroll
    for (int v = 0; v < 8; ++v) {
      float r  = 1.0f / (1.0f + __expf(-(acc_r[v] + bi[0] + bh[0])));
      float z  = 1.0f / (1.0f + __expf(-(acc_z[v] + bi[1] + bh[1])));
      float np = acc_xn[v] + bi[2] + r * (acc_hn[v] + bh[2]);
      float n  = 2.0f / (1.0f + __expf(-2.0f * np)) - 1.0f;
      float hn = (1.0f - z) * n + z * hreg[v];
      hreg[v] = hn;
      wb[(v + 8 * hi) * HP + w * 16 + lnib] = (_Float16)hn;
    }

    __syncthreads();
  }

  // final hidden state -> out [B, H] f32
  #pragma unroll
  for (int v = 0; v < 8; ++v)
    out[(size_t)(b0 + v + 8 * hi) * H_ + w * 16 + lnib] = hreg[v];
}

// ---------------------------------------------------------------------------
extern "C" void kernel_launch(void* const* d_in, const int* in_sizes, int n_in,
                              void* d_out, int out_size, void* d_ws, size_t ws_size,
                              hipStream_t stream) {
  const float* input = (const float*)d_in[0];
  const float* W_ih  = (const float*)d_in[1];
  const float* W_hh  = (const float*)d_in[2];
  const float* b_ih  = (const float*)d_in[3];
  const float* b_hh  = (const float*)d_in[4];
  _Float16* ws16 = (_Float16*)d_ws;   // needs 589824 B of workspace
  float* out = (float*)d_out;

  const int total = WIH_HALVES + WHH_HALVES;
  gru_prep<<<(total + 255) / 256, 256, 0, stream>>>(W_ih, W_hh, ws16);
  gru_scan<<<B_ / 16, 512, 0, stream>>>(input, b_ih, b_hh, ws16, out);
}
